// GatedGraphConv_7516192768588
// MI455X (gfx1250) — compile-verified, matrix operand order checked by reading, not run on hardware
//
#include <hip/hip_runtime.h>
#include <hip/hip_bf16.h>
#include <math.h>

// GatedGraphConv for MI455X (gfx1250), wave32 + WMMA bf16.
// B=32, N=100, NODE_DIM=128, EDGE_DIM=64, HID=128.
// Outputs: h_out (3200*128) then e_norm (320000*128), both f32.

typedef __attribute__((ext_vector_type(16))) __bf16 v16bf;
typedef __attribute__((ext_vector_type(8)))  float  v8f;

#define NB      32
#define NN      100
#define NDIM    128
#define EDIM    64
#define HID     128
#define ROWS_H  (NB * NN)            // 3200
#define ROWS_E  (NB * NN * NN)       // 320000
#define RT_E    (ROWS_E / 16)        // 20000 row tiles
#define NPART   512                  // edge-BN partial slots (== k_edge_gemm grid)

// Workspace layout (float offsets). Total 2,179,584 floats (~8.4 MB).
#define WS_AH    0
#define WS_BH    (WS_AH   + ROWS_H * HID)       // 409600
#define WS_HIN   (WS_BH   + ROWS_H * HID)       // 819200
#define WS_HNB   (WS_HIN  + ROWS_H * HID)       // 1228800
#define WS_PSUM  (WS_HNB  + ROWS_H * HID)       // 1638400
#define WS_PSQ   (WS_PSUM + NPART * HID)        // +65536
#define WS_ESC   (WS_PSQ  + NPART * HID)        // +65536
#define WS_ESH   (WS_ESC  + HID)
#define WS_M     (WS_ESH  + HID)
#define WS_NSC   (WS_M    + ROWS_H * HID)
#define WS_NSH   (WS_NSC  + HID)

// K index inside a 16x32 bf16 WMMA fragment (16 elems/lane, wave32):
// VGPR v half hh, lanes 0-15: K = 2v+hh (v<4), 16+... ; lanes 16-31: +8.
__device__ __forceinline__ int frag_k(int lane, int i) {
    return (i & 7) + ((lane & 16) ? 8 : 0) + ((i & 8) ? 16 : 0);
}

// ---------------------------------------------------------------------------
// K1: four node GEMMs  out[r,d] = sum_c h[r,c] * W[d,c]   (3200x128 @ 128x128)
// grid = 200 row-tiles * 4 matrices; block = 256 (8 waves, one 16-col tile each)
// ---------------------------------------------------------------------------
__global__ __launch_bounds__(256) void k_node_gemm(
    const float* __restrict__ h,
    const float* __restrict__ WU, const float* __restrict__ WV,
    const float* __restrict__ WA, const float* __restrict__ WB,
    float* __restrict__ ws)
{
    __shared__ float ldsA[16 * NDIM];
    const int tile    = blockIdx.x;
    const int matId   = tile / 200;
    const int rowTile = tile % 200;
    const int tid  = threadIdx.x;
    const int lane = tid & 31;
    const int wave = tid >> 5;

    const float* W = (matId == 0) ? WU : (matId == 1) ? WV : (matId == 2) ? WA : WB;
    float* outp = ws + ((matId == 0) ? WS_HIN : (matId == 1) ? WS_HNB
                       : (matId == 2) ? WS_AH : WS_BH);

    const int rowBase = rowTile * 16;
    for (int i = tid; i < 16 * NDIM; i += 256) ldsA[i] = h[rowBase * NDIM + i];
    __syncthreads();

    const int colBase = wave * 16;
    const int nIdx    = colBase + (lane & 15);
    const int aRow    = lane & 15;
    const int rowHalf = (lane & 16) ? 8 : 0;

    v8f c = {};
#pragma unroll
    for (int ks = 0; ks < 4; ++ks) {
        v16bf a, b;
#pragma unroll
        for (int i = 0; i < 16; ++i) {
            const int k = ks * 32 + frag_k(lane, i);
            a[i] = (__bf16)ldsA[aRow * NDIM + k];
            b[i] = (__bf16)W[nIdx * NDIM + k];
        }
        c = __builtin_amdgcn_wmma_f32_16x16x32_bf16(false, a, false, b,
                                                    (short)0, c, false, false);
    }
#pragma unroll
    for (int j = 0; j < 8; ++j) {
        const int r = rowBase + j + rowHalf;
        outp[r * HID + nIdx] = c[j];
    }
}

// ---------------------------------------------------------------------------
// K2: edge GEMM  eh[r,d] = sum_k e[r,k]*WC[d,k] + Ah[b,m,d] + Bh[b,n,d]
// r = (b*100+m)*100+n. Writes eh into the e_norm output region; accumulates
// per-workgroup BN partial sums (deterministic). B fragments (WC) live in
// registers for the whole grid-stride loop; A tile staged through LDS.
// ---------------------------------------------------------------------------
__global__ __launch_bounds__(256) void k_edge_gemm(
    const float* __restrict__ e,  const float* __restrict__ WC,
    const float* __restrict__ Ah, const float* __restrict__ Bh,
    float* __restrict__ eh, float* __restrict__ psum, float* __restrict__ psq)
{
    __shared__ float ldsA[16 * EDIM];
    __shared__ float redS[256];
    __shared__ float redQ[256];

    const int tid  = threadIdx.x;
    const int lane = tid & 31;
    const int wave = tid >> 5;
    const int colBase = wave * 16;
    const int d       = colBase + (lane & 15);
    const int aRow    = lane & 15;
    const int rowHalf = (lane & 16) ? 8 : 0;

    // Pre-build WC fragments (K = 64 -> two 32-wide K steps), kept in VGPRs.
    v16bf b0, b1;
#pragma unroll
    for (int i = 0; i < 16; ++i) {
        const int k = frag_k(lane, i);
        b0[i] = (__bf16)WC[d * EDIM + k];
        b1[i] = (__bf16)WC[d * EDIM + 32 + k];
    }

    float regS = 0.f, regQ = 0.f;

    for (int t = blockIdx.x; t < RT_E; t += gridDim.x) {
        __syncthreads();
#pragma unroll 4
        for (int i = tid; i < 16 * EDIM; i += 256) ldsA[i] = e[t * (16 * EDIM) + i];
        const int tn = t + gridDim.x;
        if (tn < RT_E)  // hint next A tile toward the caches (global_prefetch_b8)
            __builtin_prefetch(&e[tn * (16 * EDIM) + tid * 4], 0, 1);
        __syncthreads();

        v16bf a0, a1;
#pragma unroll
        for (int i = 0; i < 16; ++i) {
            const int k = frag_k(lane, i);
            a0[i] = (__bf16)ldsA[aRow * EDIM + k];
            a1[i] = (__bf16)ldsA[aRow * EDIM + 32 + k];
        }
        v8f c = {};
        c = __builtin_amdgcn_wmma_f32_16x16x32_bf16(false, a0, false, b0,
                                                    (short)0, c, false, false);
        c = __builtin_amdgcn_wmma_f32_16x16x32_bf16(false, a1, false, b1,
                                                    (short)0, c, false, false);
#pragma unroll
        for (int j = 0; j < 8; ++j) {
            const int r    = t * 16 + j + rowHalf;
            const int bb   = r / (NN * NN);
            const int rem  = r - bb * (NN * NN);
            const int mrow = rem / NN;
            const int ncol = rem - mrow * NN;
            float v = c[j] + Ah[(bb * NN + mrow) * HID + d]
                           + Bh[(bb * NN + ncol) * HID + d];
            eh[r * HID + d] = v;
            regS += v;
            regQ += v * v;
        }
    }

    redS[tid] = regS;
    redQ[tid] = regQ;
    __syncthreads();
    if ((lane & 16) == 0) {  // lanes L and L+16 hold the same channel d
        psum[blockIdx.x * HID + d] = redS[tid] + redS[tid + 16];
        psq [blockIdx.x * HID + d] = redQ[tid] + redQ[tid + 16];
    }
}

// ---------------------------------------------------------------------------
// K3: finalize edge BN -> per-channel scale/shift
// ---------------------------------------------------------------------------
__global__ __launch_bounds__(128) void k_edge_bn(
    const float* __restrict__ psum, const float* __restrict__ psq,
    const float* __restrict__ gamma, const float* __restrict__ beta,
    float* __restrict__ scale, float* __restrict__ shift)
{
    const int d = threadIdx.x;
    float s = 0.f, q = 0.f;
    for (int p = 0; p < NPART; ++p) { s += psum[p * HID + d]; q += psq[p * HID + d]; }
    const float inv  = 1.0f / (float)ROWS_E;
    const float mean = s * inv;
    const float var  = q * inv - mean * mean;
    const float sc   = gamma[d] * rsqrtf(var + 1e-5f);
    scale[d] = sc;
    shift[d] = beta[d] - mean * sc;
}

// ---------------------------------------------------------------------------
// K4: per (b,m) slab: BN -> relu -> adj mask -> sigmoid -> softmax over n
//     -> write e_norm (in place over eh) and aggregate m[b,m,d].
// One thread per channel d; 100x128 slab staged in LDS.
// ---------------------------------------------------------------------------
__global__ __launch_bounds__(128) void k_softmax_agg(
    const float* __restrict__ adj, const float* __restrict__ hnb,
    const float* __restrict__ escale, const float* __restrict__ eshift,
    float* __restrict__ enorm /* holds eh on entry */, float* __restrict__ mbuf)
{
    __shared__ float wslab[NN * HID];   // 51.2 KB
    __shared__ float adjRow[NN];

    const int bm = blockIdx.x;          // b*100 + m
    const int d  = threadIdx.x;
    const int bb = bm / NN;

    if (d < NN) adjRow[d] = adj[bm * NN + d];
    __syncthreads();

    const float sc = escale[d], sh = eshift[d];
    const float* ehrow = enorm + bm * (NN * HID);

    float wmax = -1e30f;
    for (int n = 0; n < NN; ++n) {
        const float x = ehrow[n * HID + d];
        float y = fmaxf(x * sc + sh, 0.f) * adjRow[n];
        const float w = 1.f / (1.f + __expf(-y));
        wslab[n * HID + d] = w;
        wmax = fmaxf(wmax, w);
    }

    const float* hnbB = hnb + bb * NN * HID;
    float S = 0.f, T = 0.f;
    for (int n = 0; n < NN; ++n) {
        const float s = __expf(wslab[n * HID + d] - wmax);
        wslab[n * HID + d] = s;
        S += s;
        T += s * hnbB[n * HID + d];
    }
    const float invS = 1.f / S;

    float* outrow = enorm + bm * (NN * HID);
    for (int n = 0; n < NN; ++n) outrow[n * HID + d] = wslab[n * HID + d] * invS;

    mbuf[bm * HID + d] = T * invS;
}

// ---------------------------------------------------------------------------
// K5: node BN stats over m (3200 rows), deterministic single workgroup
// ---------------------------------------------------------------------------
__global__ __launch_bounds__(128) void k_node_bn(
    const float* __restrict__ mbuf,
    const float* __restrict__ gamma, const float* __restrict__ beta,
    float* __restrict__ scale, float* __restrict__ shift)
{
    const int d = threadIdx.x;
    float s = 0.f, q = 0.f;
    for (int r = 0; r < ROWS_H; ++r) {
        const float v = mbuf[r * HID + d];
        s += v; q += v * v;
    }
    const float inv  = 1.0f / (float)ROWS_H;
    const float mean = s * inv;
    const float var  = q * inv - mean * mean;
    const float sc   = gamma[d] * rsqrtf(var + 1e-5f);
    scale[d] = sc;
    shift[d] = beta[d] - mean * sc;
}

// ---------------------------------------------------------------------------
// K6: h_out = h_in + relu(bn(m))
// ---------------------------------------------------------------------------
__global__ __launch_bounds__(256) void k_node_out(
    const float* __restrict__ hin, const float* __restrict__ mbuf,
    const float* __restrict__ scale, const float* __restrict__ shift,
    float* __restrict__ hout)
{
    const int i = blockIdx.x * 256 + threadIdx.x;
    if (i >= ROWS_H * HID) return;
    const int d = i & (HID - 1);
    const float mm = mbuf[i] * scale[d] + shift[d];
    hout[i] = hin[i] + fmaxf(mm, 0.f);
}

// ---------------------------------------------------------------------------
extern "C" void kernel_launch(void* const* d_in, const int* in_sizes, int n_in,
                              void* d_out, int out_size, void* d_ws, size_t ws_size,
                              hipStream_t stream)
{
    const float* h   = (const float*)d_in[0];
    const float* e   = (const float*)d_in[1];
    const float* adj = (const float*)d_in[2];
    const float* WU  = (const float*)d_in[3];
    const float* WV  = (const float*)d_in[4];
    const float* WA  = (const float*)d_in[5];
    const float* WB  = (const float*)d_in[6];
    const float* WC  = (const float*)d_in[7];
    const float* eg  = (const float*)d_in[8];
    const float* eb  = (const float*)d_in[9];
    const float* ng  = (const float*)d_in[10];
    const float* nb  = (const float*)d_in[11];

    float* out   = (float*)d_out;
    float* hout  = out;                       // 3200*128
    float* enorm = out + ROWS_H * HID;        // 320000*128 (also scratch for eh)

    float* ws   = (float*)d_ws;               // needs ~8.4 MB
    float* Ah   = ws + WS_AH;
    float* Bh   = ws + WS_BH;
    float* hin  = ws + WS_HIN;
    float* hnb  = ws + WS_HNB;
    float* psum = ws + WS_PSUM;
    float* psq  = ws + WS_PSQ;
    float* esc  = ws + WS_ESC;
    float* esh  = ws + WS_ESH;
    float* mbuf = ws + WS_M;
    float* nsc  = ws + WS_NSC;
    float* nsh  = ws + WS_NSH;

    k_node_gemm  <<<200 * 4, 256, 0, stream>>>(h, WU, WV, WA, WB, ws);
    k_edge_gemm  <<<NPART,   256, 0, stream>>>(e, WC, Ah, Bh, enorm, psum, psq);
    k_edge_bn    <<<1,       128, 0, stream>>>(psum, psq, eg, eb, esc, esh);
    k_softmax_agg<<<ROWS_H,  128, 0, stream>>>(adj, hnb, esc, esh, enorm, mbuf);
    k_node_bn    <<<1,       128, 0, stream>>>(mbuf, ng, nb, nsc, nsh);
    k_node_out   <<<(ROWS_H * HID + 255) / 256, 256, 0, stream>>>(hin, mbuf, nsc, nsh, hout);
}
